// GAT_15625091022894
// MI455X (gfx1250) — matrix operand order repaired and checked by reading, not compiled, not run on hardware
//
#include <hip/hip_runtime.h>
#include <math.h>

#define N_NODES 50000
#define N_EDGES 1600000
#define CLS 40
#define SLOPE 0.2f

typedef __attribute__((ext_vector_type(16))) _Float16 v16h;
typedef __attribute__((ext_vector_type(8)))  float    v8f;

// ---------------- generic fill ----------------
__global__ void fill_kernel(float* __restrict__ p, int n, float v) {
    int i = blockIdx.x * blockDim.x + threadIdx.x;
    if (i < n) p[i] = v;
}

// ---------------- GEMM: C[M x Nc] = A[M x 128] * B[128 x Nc] -----------------
// f16 WMMA with f32 accumulation. Block tile 64(M) x 32(N), K=128 staged fully
// in LDS. 8 waves per block, each wave owns one 16x16 output tile.
__global__ __launch_bounds__(256)
void gemm_wmma(const float* __restrict__ A, const float* __restrict__ B,
               float* __restrict__ C, int M, int Nc) {
    constexpr int LDA = 132;                 // halfs per LDS row (pad vs. conflicts)
    __shared__ _Float16 As[64 * LDA];        // [row][k]      64 x 128
    __shared__ _Float16 Bs[32 * LDA];        // [col][k]  B transposed, 32 x 128

    const int t  = threadIdx.x;
    const int bm = blockIdx.x * 64;
    const int bn = blockIdx.y * 32;
    const bool m_full = (bm + 64 <= M);      // uniform: 781/782 blocks take fast path

    // Load A tile 64x128 (float4, coalesced), convert to f16.
    if (m_full) {
        #pragma unroll
        for (int j = 0; j < 8; ++j) {
            int q = t + 256 * j;             // 2048 float4 total
            int r = (q * 4) / 128, c = (q * 4) % 128;
            float4 fv = *(const float4*)(A + (size_t)(bm + r) * 128 + c);
            _Float16* d = &As[r * LDA + c];
            d[0] = (_Float16)fv.x; d[1] = (_Float16)fv.y;
            d[2] = (_Float16)fv.z; d[3] = (_Float16)fv.w;
        }
    } else {
        #pragma unroll
        for (int j = 0; j < 8; ++j) {
            int q = t + 256 * j;
            int r = (q * 4) / 128, c = (q * 4) % 128;
            float4 fv = make_float4(0.f, 0.f, 0.f, 0.f);
            if (bm + r < M) fv = *(const float4*)(A + (size_t)(bm + r) * 128 + c);
            _Float16* d = &As[r * LDA + c];
            d[0] = (_Float16)fv.x; d[1] = (_Float16)fv.y;
            d[2] = (_Float16)fv.z; d[3] = (_Float16)fv.w;
        }
    }
    // Load B tile 128x32 (coalesced on columns), store transposed [col][k].
    #pragma unroll
    for (int j = 0; j < 16; ++j) {
        int q = t + 256 * j;                 // 4096 elements
        int k = q / 32, c = q % 32;
        int gn = bn + c;
        float fv = (gn < Nc) ? B[(size_t)k * Nc + gn] : 0.f;
        Bs[c * LDA + k] = (_Float16)fv;
    }
    __syncthreads();

    const int lane = t & 31;
    const int wave = t >> 5;
    const int wm   = (wave & 3) * 16;        // 0,16,32,48
    const int wn   = (wave >> 2) * 16;       // 0,16
    const int lm   = lane & 15;
    const int hi   = (lane >> 4) & 1;

    v8f acc = {};
    const _Float16* arow = &As[(wm + lm) * LDA];   // A: M = lane%16
    const _Float16* brow = &Bs[(wn + lm) * LDA];   // B: N = lane%16
    #pragma unroll
    for (int kk = 0; kk < 128; kk += 32) {
        v16h a, b;
        #pragma unroll
        for (int vv = 0; vv < 8; ++vv) {
            // A 16x32 f16, contiguous k order chosen here (same as B): lanes 0-15 K=0..15, lanes 16-31 K=16..31
            int ka = kk + hi * 16 + 2 * vv;
            a[2 * vv]     = arow[ka];
            a[2 * vv + 1] = arow[ka + 1];
            // B 32x16 f16, contiguous k order chosen here (same as A): lanes 0-15 K=0..15, lanes 16-31 K=16..31
            int kb = kk + hi * 16 + 2 * vv;
            b[2 * vv]     = brow[kb];
            b[2 * vv + 1] = brow[kb + 1];
        }
        acc = __builtin_amdgcn_wmma_f32_16x16x32_f16(false, a, false, b,
                                                     (short)0, acc, false, false);
    }
    // D layout: VGPR v -> M = row0 + v (row0 = +8 for lanes 16-31), N = lane%16.
    // Rows per lane are contiguous: walk one base pointer, no 64-bit multiplies.
    const int col  = bn + wn + lm;
    const int row0 = bm + wm + hi * 8;
    if (col < Nc) {
        float* cp = C + (size_t)row0 * Nc + col;
        if (m_full) {
            #pragma unroll
            for (int vv = 0; vv < 8; ++vv) { *cp = acc[vv]; cp += Nc; }
        } else {
            #pragma unroll
            for (int vv = 0; vv < 8; ++vv) {
                if (row0 + vv < M) *cp = acc[vv];
                cp += Nc;
            }
        }
    }
}

// ---------------- per-node attention coefficients ----------------
__global__ void el_er_kernel(const float* __restrict__ f,
                             const float* __restrict__ al,
                             const float* __restrict__ ar,
                             float* __restrict__ el, float* __restrict__ er,
                             int H, int D) {
    int i = blockIdx.x * blockDim.x + threadIdx.x;   // node*H + h
    if (i >= N_NODES * H) return;
    int h = i % H;
    const float* fp  = f + (size_t)i * D;            // f[(n*H+h)*D]
    const float* alp = al + h * D;
    const float* arp = ar + h * D;
    float sl = 0.f, sr = 0.f;
    for (int d = 0; d < D; ++d) { float x = fp[d]; sl += x * alp[d]; sr += x * arp[d]; }
    el[i] = sl; er[i] = sr;
}

// ---------------- edge pass 1: logits + segment max ----------------
__global__ void edge_logits_kernel(const int* __restrict__ src, const int* __restrict__ dst,
                                   const float* __restrict__ el, const float* __restrict__ er,
                                   float* __restrict__ e, float* __restrict__ m, int H) {
    int i = blockIdx.x * blockDim.x + threadIdx.x;
    if (i >= N_EDGES) return;
    int s_ = src[i], d_ = dst[i];
    for (int h = 0; h < H; ++h) {
        float v  = el[s_ * H + h] + er[d_ * H + h];
        float lr = v > 0.f ? v : SLOPE * v;
        e[(size_t)i * H + h] = lr;
        float* addr = &m[d_ * H + h];
        // float max via int ordering trick (works across signs, init = -inf)
        if (lr >= 0.f) atomicMax((int*)addr, __float_as_int(lr));
        else           atomicMin((unsigned int*)addr, __float_as_uint(lr));
    }
}

// ---------------- edge pass 2: exp + segment sum ----------------
__global__ void edge_exp_kernel(const int* __restrict__ dst,
                                const float* __restrict__ m,
                                float* __restrict__ e, float* __restrict__ ssum, int H) {
    int i = blockIdx.x * blockDim.x + threadIdx.x;
    if (i >= N_EDGES) return;
    int d_ = dst[i];
    for (int h = 0; h < H; ++h) {
        float mm = m[d_ * H + h];
        if (mm == -INFINITY) mm = 0.f;               // matches where(isfinite(m), m, 0)
        float ee = __expf(e[(size_t)i * H + h] - mm);
        e[(size_t)i * H + h] = ee;
        atomicAdd(&ssum[d_ * H + h], ee);
    }
}

// ---------------- edge pass 3: alpha-weighted aggregation ----------------
// One wave32 per edge; iteration h covers head h's D contiguous features.
__global__ void edge_agg_kernel(const int* __restrict__ src, const int* __restrict__ dst,
                                const float* __restrict__ e, const float* __restrict__ ssum,
                                const float* __restrict__ f, float* __restrict__ out,
                                int H, int D) {
    int w    = (blockIdx.x * blockDim.x + threadIdx.x) >> 5;
    int lane = threadIdx.x & 31;
    if (w >= N_EDGES) return;
    int s_ = src[w], d_ = dst[w];
    int F = H * D;
    for (int h = 0; h < H; ++h) {
        float alpha = e[(size_t)w * H + h] / ssum[d_ * H + h];
        int base = h * D;
        for (int t2 = lane; t2 < D; t2 += 32) {
            atomicAdd(&out[(size_t)d_ * F + base + t2],
                      alpha * f[(size_t)s_ * F + base + t2]);
        }
    }
}

// ---------------- ELU ----------------
__global__ void elu_kernel(float* __restrict__ x, int n) {
    int i = blockIdx.x * blockDim.x + threadIdx.x;
    if (i >= n) return;
    float v = x[i];
    x[i] = v > 0.f ? v : (__expf(v) - 1.f);
}

// ---------------- log_softmax over CLS classes ----------------
__global__ void logsoftmax_kernel(const float* __restrict__ logits, float* __restrict__ out) {
    int n = blockIdx.x * blockDim.x + threadIdx.x;
    if (n >= N_NODES) return;
    const float* p = logits + (size_t)n * CLS;
    float mx = -INFINITY;
    for (int c = 0; c < CLS; ++c) mx = fmaxf(mx, p[c]);
    float s = 0.f;
    for (int c = 0; c < CLS; ++c) s += __expf(p[c] - mx);
    float lse = mx + __logf(s);
    float* o = out + (size_t)n * CLS;
    for (int c = 0; c < CLS; ++c) o[c] = p[c] - lse;
}

extern "C" void kernel_launch(void* const* d_in, const int* in_sizes, int n_in,
                              void* d_out, int out_size, void* d_ws, size_t ws_size,
                              hipStream_t stream) {
    const float* X   = (const float*)d_in[0];
    const int*   src = (const int*)  d_in[1];
    const int*   dst = (const int*)  d_in[2];
    const float* W0  = (const float*)d_in[3];
    const float* al0 = (const float*)d_in[4];
    const float* ar0 = (const float*)d_in[5];
    const float* W1  = (const float*)d_in[6];
    const float* al1 = (const float*)d_in[7];
    const float* ar1 = (const float*)d_in[8];
    const float* W2  = (const float*)d_in[9];
    const float* al2 = (const float*)d_in[10];
    const float* ar2 = (const float*)d_in[11];

    float* out  = (float*)d_out;                 // [N, CLS] log-softmax
    float* hout = out + (size_t)N_NODES * CLS;   // [N, 128] layer-1 activations (2nd output)

    // workspace layout (floats): ~88 MB
    float* ws   = (float*)d_ws;
    float* f    = ws;                                  // N*128 projected features
    float* x1   = f    + (size_t)N_NODES * 128;        // N*128 layer-0 output
    float* ebuf = x1   + (size_t)N_NODES * 128;        // E*4   edge logits / exp
    float* el   = ebuf + (size_t)N_EDGES * 4;          // N*4
    float* er   = el   + (size_t)N_NODES * 4;          // N*4
    float* m    = er   + (size_t)N_NODES * 4;          // N*4
    float* ssum = m    + (size_t)N_NODES * 4;          // N*4
    float* out2 = ssum + (size_t)N_NODES * 4;          // N*CLS layer-2 logits

    auto run_layer = [&](const float* xin, const float* W, const float* al, const float* ar,
                         float* fl, float* yout, int H, int D, bool act) {
        int F  = H * D;
        int nh = N_NODES * H;
        int nf = N_NODES * F;
        dim3 gg((N_NODES + 63) / 64, (F + 31) / 32);
        gemm_wmma<<<gg, 256, 0, stream>>>(xin, W, fl, N_NODES, F);
        el_er_kernel<<<(nh + 255) / 256, 256, 0, stream>>>(fl, al, ar, el, er, H, D);
        fill_kernel<<<(nh + 255) / 256, 256, 0, stream>>>(m, nh, -INFINITY);
        fill_kernel<<<(nh + 255) / 256, 256, 0, stream>>>(ssum, nh, 0.f);
        fill_kernel<<<(nf + 255) / 256, 256, 0, stream>>>(yout, nf, 0.f);
        edge_logits_kernel<<<(N_EDGES + 255) / 256, 256, 0, stream>>>(src, dst, el, er, ebuf, m, H);
        edge_exp_kernel<<<(N_EDGES + 255) / 256, 256, 0, stream>>>(dst, m, ebuf, ssum, H);
        long wthreads = (long)N_EDGES * 32;
        edge_agg_kernel<<<(unsigned)((wthreads + 255) / 256), 256, 0, stream>>>(
            src, dst, ebuf, ssum, fl, yout, H, D);
        if (act) elu_kernel<<<(nf + 255) / 256, 256, 0, stream>>>(yout, nf);
    };

    run_layer(X,    W0, al0, ar0, f, x1,   4, 32, true);   // layer 0 -> x1
    run_layer(x1,   W1, al1, ar1, f, hout, 4, 32, true);   // layer 1 -> h (d_out region)
    run_layer(hout, W2, al2, ar2, f, out2, 1, 40, false);  // layer 2 -> logits

    logsoftmax_kernel<<<(N_NODES + 255) / 256, 256, 0, stream>>>(out2, out);
}
